// Critic_60516089200892
// MI455X (gfx1250) — matrix operand order repaired and checked by reading, not compiled, not run on hardware
//
#include <hip/hip_runtime.h>
#include <hip/hip_bf16.h>

#define BDIM 4096
#define TDIM 64
#define HDIM 64
#define NROWS (BDIM * TDIM)

typedef __attribute__((ext_vector_type(16))) _Float16 v16h;
typedef __attribute__((ext_vector_type(8)))  _Float16 v8h;
typedef __attribute__((ext_vector_type(8)))  float    v8f;

static __device__ inline v8f wmma_f16(v16h a, v16h b, v8f c) {
  // D = A(16x32 f16) * B(32x16 f16) + C(16x16 f32)
  return __builtin_amdgcn_wmma_f32_16x16x32_f16(false, a, false, b, (short)0, c, false, false);
}

// A-fragment (16x32, f16) from a row-major f16 tile (LDS or global).
// Layout: lanes 0-15 -> M=0..15, K=k0+0..15; lanes 16-31 -> same M, K=k0+16..31 (contiguous, matches B)
static __device__ inline v16h loadA_h(const _Float16* base, int m, int k0, int stride, int lane) {
  const _Float16* p = base + (size_t)m * stride + k0 + 16 * (lane >> 4);
  v8h lo = *(const v8h*)p;
  v8h hi = *(const v8h*)(p + 8);
  v16h r;
#pragma unroll
  for (int e = 0; e < 8; e++) { r[e] = lo[e]; r[e + 8] = hi[e]; }
  return r;
}

// A-fragment from a row-major f32 row (global), converting to f16.
static __device__ inline v16h loadA_f32(const float* row, int k0, int lane) {
  const float* p = row + k0 + 16 * (lane >> 4);
  v16h r;
#pragma unroll
  for (int e = 0; e < 8; e++) { r[e] = (_Float16)p[e]; r[e + 8] = (_Float16)p[8 + e]; }
  return r;
}

// B-fragment (32x16, f16) from an N-major (transposed) f16 weight tile WT[n][k].
// Layout: lane n = n0+(lane&15); lanes 0-15 hold K=k0..k0+15, lanes 16-31 hold K=k0+16..k0+31 (contiguous 16 halfs)
static __device__ inline v16h loadB_h(const _Float16* wt, int n0, int k0, int kstride, int lane) {
  const _Float16* p = wt + (size_t)(n0 + (lane & 15)) * kstride + k0 + 16 * (lane >> 4);
  return *(const v16h*)p;
}

// B-fragment gathered straight from a K-major f32 global weight W[k][256] (coalesced across lanes per element).
static __device__ inline v16h loadB_f32T(const float* W, int krow0, int n, int k0, int lane) {
  int kb = krow0 + k0 + 16 * (lane >> 4);
  v16h r;
#pragma unroll
  for (int e = 0; e < 16; e++) r[e] = (_Float16)W[(size_t)(kb + e) * 256 + n];
  return r;
}

static __device__ inline float sigmoidf_(float x) { return 1.0f / (1.0f + __expf(-x)); }

// LayerNorm(64) + ReLU on 4 C-tiles in WMMA C layout: lane holds (m = e + 8*(lane>=16), n = t*16 + (lane&15))
static __device__ inline void ln_relu4(v8f acc[4], const float* g, const float* be, int lane) {
  v8f s = acc[0] + acc[1] + acc[2] + acc[3];
  v8f q = acc[0] * acc[0] + acc[1] * acc[1] + acc[2] * acc[2] + acc[3] * acc[3];
#pragma unroll
  for (int mask = 1; mask < 16; mask <<= 1) {
#pragma unroll
    for (int e = 0; e < 8; e++) {
      s[e] += __shfl_xor(s[e], mask, 32);
      q[e] += __shfl_xor(q[e], mask, 32);
    }
  }
  int nl = lane & 15;
  float gv[4], bv[4];
#pragma unroll
  for (int t = 0; t < 4; t++) { gv[t] = g[t * 16 + nl]; bv[t] = be[t * 16 + nl]; }
#pragma unroll
  for (int e = 0; e < 8; e++) {
    float mean = s[e] * (1.0f / 64.0f);
    float var  = q[e] * (1.0f / 64.0f) - mean * mean;
    float r = rsqrtf(var + 1e-12f);
#pragma unroll
    for (int t = 0; t < 4; t++) {
      float v = (acc[t][e] - mean) * r * gv[t] + bv[t];
      acc[t][e] = v > 0.0f ? v : 0.0f;
    }
  }
}

// ---------------- Kernel 1: MLP trunk (two GEMM-LN-ReLU layers), writes x as f16 ----------------
// block: 256 threads (8 waves), 128 rows/block; wave w owns rows [w*16, w*16+16), all 64 output cols.
__global__ void __launch_bounds__(256) trunk_kernel(
    const float* __restrict__ obs, const float* __restrict__ act,
    const float* __restrict__ W1, const float* __restrict__ b1,
    const float* __restrict__ g1, const float* __restrict__ be1,
    const float* __restrict__ W2, const float* __restrict__ b2,
    const float* __restrict__ g2, const float* __restrict__ be2,
    _Float16* __restrict__ xout) {
  __shared__ __align__(32) _Float16 sW1T[64 * 128];  // W1T[n][k], 16 KB
  __shared__ __align__(32) _Float16 sW2T[64 * 96];   // W2T[n][k], 12 KB
  __shared__ __align__(32) _Float16 sX[128 * 64];    // per-block activation tile, 16 KB

  int tid = threadIdx.x, lane = tid & 31, w = tid >> 5;
  for (int i = tid; i < 64 * 128; i += 256) { int n = i >> 7, k = i & 127; sW1T[i] = (_Float16)W1[k * 64 + n]; }
  for (int i = tid; i < 64 * 96; i += 256) { int n = i / 96, k = i % 96; sW2T[i] = (_Float16)W2[k * 64 + n]; }
  __syncthreads();

  int m = lane & 15, hi = lane >> 4, nl = lane & 15;
  size_t grow = (size_t)blockIdx.x * 128 + w * 16 + m;
  const float* orow = obs + grow * 128;

  // ---- layer 1: [16,128] @ [128,64] ; hoist all fragments so loads pipeline, WMMAs go back-to-back
  v16h A1[4], B1f[4][4];
#pragma unroll
  for (int ks = 0; ks < 4; ks++) {
    A1[ks] = loadA_f32(orow, ks * 32, lane);
#pragma unroll
    for (int t = 0; t < 4; t++) B1f[ks][t] = loadB_h(sW1T, t * 16, ks * 32, 128, lane);
  }
  v8f acc[4];
#pragma unroll
  for (int t = 0; t < 4; t++) { float bv = b1[t * 16 + nl];
#pragma unroll
    for (int e = 0; e < 8; e++) acc[t][e] = bv; }
#pragma unroll
  for (int ks = 0; ks < 4; ks++)
#pragma unroll
    for (int t = 0; t < 4; t++) acc[t] = wmma_f16(A1[ks], B1f[ks][t], acc[t]);
  ln_relu4(acc, g1, be1, lane);

  // stage x1 in LDS (wave-local rows; LDS ops are in-order per wave)
#pragma unroll
  for (int t = 0; t < 4; t++)
#pragma unroll
    for (int e = 0; e < 8; e++)
      sX[(w * 16 + e + 8 * hi) * 64 + t * 16 + nl] = (_Float16)acc[t][e];

  // ---- layer 2: [16,96] @ [96,64], K = 64 (x1 from LDS) + 32 (action from global)
  v16h A2[3], B2f[3][4];
  A2[0] = loadA_h(sX + (size_t)w * 16 * 64, m, 0, 64, lane);
  A2[1] = loadA_h(sX + (size_t)w * 16 * 64, m, 32, 64, lane);
  A2[2] = loadA_f32(act + grow * 32, 0, lane);
#pragma unroll
  for (int ks = 0; ks < 3; ks++)
#pragma unroll
    for (int t = 0; t < 4; t++) B2f[ks][t] = loadB_h(sW2T, t * 16, ks * 32, 96, lane);
  v8f acc2[4];
#pragma unroll
  for (int t = 0; t < 4; t++) { float bv = b2[t * 16 + nl];
#pragma unroll
    for (int e = 0; e < 8; e++) acc2[t][e] = bv; }
#pragma unroll
  for (int ks = 0; ks < 3; ks++)
#pragma unroll
    for (int t = 0; t < 4; t++) acc2[t] = wmma_f16(A2[ks], B2f[ks][t], acc2[t]);
  ln_relu4(acc2, g2, be2, lane);

  // stage x2 and copy out coalesced
#pragma unroll
  for (int t = 0; t < 4; t++)
#pragma unroll
    for (int e = 0; e < 8; e++)
      sX[(w * 16 + e + 8 * hi) * 64 + t * 16 + nl] = (_Float16)acc2[t][e];
  __syncthreads();
  const uint32_t* src = (const uint32_t*)sX;
  uint32_t* dst = (uint32_t*)(xout + (size_t)blockIdx.x * 128 * 64);
  for (int i = tid; i < 4096; i += 256) dst[i] = src[i];
}

// ---------------- Kernel 2: bidirectional LSTM scan, fused h.wx / h.wp readout ----------------
// grid (B/32, 2): block owns 32 sequences for one direction; 8 waves:
//   wave = (msub = w>>2 : row half, wq = w&3 : hidden col group of 16 across all four gates)
__global__ void __launch_bounds__(256) lstm_kernel(
    const _Float16* __restrict__ xws,
    const float* __restrict__ Wf, const float* __restrict__ bf,
    const float* __restrict__ Wb, const float* __restrict__ bb,
    const float* __restrict__ wx, const float* __restrict__ bx,
    const float* __restrict__ wp, const float* __restrict__ bp,
    float* __restrict__ xs, float* __restrict__ ps) {
  __shared__ __align__(32) _Float16 sH[32 * 64];   // h tile for A-fragment relayout (4 KB)
  __shared__ float sXSp[4][32], sPSp[4][32];       // per-wq deterministic partial sums

  int dir = blockIdx.y;
  const float* W  = dir ? Wb : Wf;
  const float* bg = dir ? bb : bf;

  int tid = threadIdx.x, lane = tid & 31, w = tid >> 5;
  int msub = w >> 2, wq = w & 3;
  int m = lane & 15, hi = lane >> 4;
  int n = wq * 16 + m;                 // this lane's hidden column
  size_t bBase = (size_t)blockIdx.x * 32;

  // Loop-invariant B fragments straight from global f32 weights (coalesced gathers, kept in VGPRs)
  v16h Bx[4][2], Bh[4][2];
#pragma unroll
  for (int g = 0; g < 4; g++)
#pragma unroll
    for (int ks = 0; ks < 2; ks++) {
      Bx[g][ks] = loadB_f32T(W, 0,  g * 64 + n, ks * 32, lane);  // x-part rows 0..63
      Bh[g][ks] = loadB_f32T(W, 64, g * 64 + n, ks * 32, lane);  // h-part rows 64..127
    }

  float bgv[4];
#pragma unroll
  for (int g = 0; g < 4; g++) bgv[g] = bg[64 * g + n];
  float wxv = wx[n], wpv = wp[n];
  float bxv = bx[0], bpv = bp[0];

  for (int i = tid; i < 32 * 64; i += 256) sH[i] = (_Float16)0.0f;  // h0 = 0
  __syncthreads();

  v8f c;
#pragma unroll
  for (int e = 0; e < 8; e++) c[e] = 0.0f;

  const _Float16* xseq = xws + (bBase + msub * 16 + m) * TDIM * 64;  // this lane's sequence base

  for (int tt = 0; tt < TDIM; tt++) {
    int t = dir ? (TDIM - 1 - tt) : tt;
    // A fragments: x_t (global f16, row b*T+t) and h_{t-1} (LDS)
    const _Float16* xrow = xseq + (size_t)t * 64;
    v16h ax0 = loadA_h(xrow, 0, 0, 64, lane);
    v16h ax1 = loadA_h(xrow, 0, 32, 64, lane);
    v16h ah0 = loadA_h(sH + (size_t)msub * 16 * 64, m, 0, 64, lane);
    v16h ah1 = loadA_h(sH + (size_t)msub * 16 * 64, m, 32, 64, lane);

    // prefetch next timestep's x row while this step computes (global_prefetch_b8)
    if (tt + 1 < TDIM) {
      int tn = dir ? (TDIM - 2 - tt) : (tt + 1);
      __builtin_prefetch(xseq + (size_t)tn * 64, 0, 3);
    }

    v8f z[4];
#pragma unroll
    for (int g = 0; g < 4; g++) {
      v8f a;
#pragma unroll
      for (int e = 0; e < 8; e++) a[e] = bgv[g];
      a = wmma_f16(ax0, Bx[g][0], a);
      a = wmma_f16(ax1, Bx[g][1], a);
      a = wmma_f16(ah0, Bh[g][0], a);
      a = wmma_f16(ah1, Bh[g][1], a);
      z[g] = a;
    }
    // gates: i=z0, j=z1, f=z2, o=z3 ; forget_bias = 1.0
    v8f h;
#pragma unroll
    for (int e = 0; e < 8; e++) {
      float ig = sigmoidf_(z[0][e]);
      float jg = tanhf(z[1][e]);
      float fg = sigmoidf_(z[2][e] + 1.0f);
      float og = sigmoidf_(z[3][e]);
      float cn = c[e] * fg + ig * jg;
      c[e] = cn;
      h[e] = tanhf(cn) * og;
    }

    // partial dots h.wx, h.wp over this wave's 16 hidden cols (rows stay per-element)
    v8f hx, hp;
#pragma unroll
    for (int e = 0; e < 8; e++) { hx[e] = h[e] * wxv; hp[e] = h[e] * wpv; }
#pragma unroll
    for (int mask = 1; mask < 16; mask <<= 1)
#pragma unroll
      for (int e = 0; e < 8; e++) { hx[e] += __shfl_xor(hx[e], mask, 32); hp[e] += __shfl_xor(hp[e], mask, 32); }

    __syncthreads();  // B1: everyone done reading sH from previous step
#pragma unroll
    for (int e = 0; e < 8; e++)
      sH[(msub * 16 + e + 8 * hi) * 64 + n] = (_Float16)h[e];
    if (m == 0) {
#pragma unroll
      for (int e = 0; e < 8; e++) {
        int rr = msub * 16 + e + 8 * hi;
        sXSp[wq][rr] = hx[e];
        sPSp[wq][rr] = hp[e];
      }
    }
    __syncthreads();  // B2: new h + partials visible

    if (tid < 32) {
      float sx = sXSp[0][tid] + sXSp[1][tid] + sXSp[2][tid] + sXSp[3][tid];
      float sp = sPSp[0][tid] + sPSp[1][tid] + sPSp[2][tid] + sPSp[3][tid];
      size_t orow = (size_t)dir * BDIM + bBase + tid;
      xs[orow * TDIM + t] = sx + bxv;
      ps[orow * TDIM + t] = sp + bpv;
    }
  }
}

// ---------------- Kernel 3: attention readout (one wave per sequence-row) ----------------
__global__ void __launch_bounds__(256) readout_kernel(
    const float* __restrict__ xs, const float* __restrict__ ps,
    const float* __restrict__ gp, const float* __restrict__ bep,
    const float* __restrict__ W3, const float* __restrict__ b3,
    float* __restrict__ out) {
  int lane = threadIdx.x & 31, wv = threadIdx.x >> 5;
  int row = blockIdx.x * 8 + wv;
  const float* psr = ps + (size_t)row * TDIM;
  const float* xsr = xs + (size_t)row * TDIM;
  float p0 = psr[lane], p1 = psr[lane + 32];

  float s = p0 + p1, q = p0 * p0 + p1 * p1;
#pragma unroll
  for (int mask = 1; mask < 32; mask <<= 1) { s += __shfl_xor(s, mask, 32); q += __shfl_xor(q, mask, 32); }
  float mean = s * (1.0f / 64.0f), var = q * (1.0f / 64.0f) - mean * mean;
  float r = rsqrtf(var + 1e-12f);
  p0 = (p0 - mean) * r * gp[lane] + bep[lane];
  p1 = (p1 - mean) * r * gp[lane + 32] + bep[lane + 32];
  p0 = p0 > 0.0f ? p0 : 0.0f;
  p1 = p1 > 0.0f ? p1 : 0.0f;

  float q0 = b3[lane], q1 = b3[lane + 32];
  for (int t = 0; t < 32; t++) {
    float pv = __shfl(p0, t, 32);
    q0 += pv * W3[t * 64 + lane];
    q1 += pv * W3[t * 64 + lane + 32];
  }
  for (int t = 0; t < 32; t++) {
    float pv = __shfl(p1, t, 32);
    q0 += pv * W3[(32 + t) * 64 + lane];
    q1 += pv * W3[(32 + t) * 64 + lane + 32];
  }
  float mx = fmaxf(q0, q1);
#pragma unroll
  for (int mask = 1; mask < 32; mask <<= 1) mx = fmaxf(mx, __shfl_xor(mx, mask, 32));
  float e0 = __expf(q0 - mx), e1 = __expf(q1 - mx);
  float den = e0 + e1;
  float acc = e0 * xsr[lane] + e1 * xsr[lane + 32];
#pragma unroll
  for (int mask = 1; mask < 32; mask <<= 1) { den += __shfl_xor(den, mask, 32); acc += __shfl_xor(acc, mask, 32); }
  if (lane == 0) out[row] = acc / den;
}

extern "C" void kernel_launch(void* const* d_in, const int* in_sizes, int n_in,
                              void* d_out, int out_size, void* d_ws, size_t ws_size,
                              hipStream_t stream) {
  const float* obs    = (const float*)d_in[0];
  const float* action = (const float*)d_in[1];
  const float* W1  = (const float*)d_in[2];
  const float* b1  = (const float*)d_in[3];
  const float* g1  = (const float*)d_in[4];
  const float* be1 = (const float*)d_in[5];
  const float* W2  = (const float*)d_in[6];
  const float* b2  = (const float*)d_in[7];
  const float* g2  = (const float*)d_in[8];
  const float* be2 = (const float*)d_in[9];
  const float* Wf  = (const float*)d_in[10];
  const float* bf  = (const float*)d_in[11];
  const float* Wb  = (const float*)d_in[12];
  const float* bb  = (const float*)d_in[13];
  const float* wx  = (const float*)d_in[14];
  const float* bx  = (const float*)d_in[15];
  const float* wp  = (const float*)d_in[16];
  const float* bp  = (const float*)d_in[17];
  const float* gp  = (const float*)d_in[18];
  const float* bep = (const float*)d_in[19];
  const float* W3  = (const float*)d_in[20];
  const float* b3  = (const float*)d_in[21];

  _Float16* xws = (_Float16*)d_ws;                                   // [N, 64] f16 : 32 MB
  float* xsb = (float*)((char*)d_ws + (size_t)NROWS * 64 * sizeof(_Float16));
  float* psb = xsb + (size_t)2 * BDIM * TDIM;                        // each [2B, T] f32 : 2 MB

  trunk_kernel<<<NROWS / 128, 256, 0, stream>>>(obs, action, W1, b1, g1, be1, W2, b2, g2, be2, xws);
  lstm_kernel<<<dim3(BDIM / 32, 2), 256, 0, stream>>>(xws, Wf, bf, Wb, bb, wx, bx, wp, bp, xsb, psb);
  readout_kernel<<<(2 * BDIM) / 8, 256, 0, stream>>>(xsb, psb, gp, bep, W3, b3, (float*)d_out);
}
